// CharDecoder_34626026340396
// MI455X (gfx1250) — matrix operand order repaired and checked by reading, not compiled, not run on hardware
//
#include <hip/hip_runtime.h>
#include <hip/hip_bf16.h>
#include <math.h>

// ---------------------------------------------------------------------------
// Problem dims (compile-time)
// ---------------------------------------------------------------------------
#define LL 48
#define BB 8192
#define EE 50
#define HH 512
#define VV 128
#define KCAT 576            // 512 (h) + 64 (x padded), = 18 * 32
#define FOURH 2048
#define LDST 40             // padded LDS row stride (bf16 elems) -> conflict-free b128

#define AS1 __attribute__((address_space(1)))
#define AS3 __attribute__((address_space(3)))

#if defined(__HIP_DEVICE_COMPILE__) && __has_builtin(__builtin_amdgcn_global_load_async_to_lds_b128)
#define USE_ASYNC_LDS 1
#else
#define USE_ASYNC_LDS 0
#endif

typedef __attribute__((ext_vector_type(8)))  float   v8f;
typedef __attribute__((ext_vector_type(8)))  __bf16  v8bf;
typedef __attribute__((ext_vector_type(16))) __bf16  v16bf;
typedef int vsi4 __attribute__((vector_size(4 * sizeof(int))));   // matches builtin param

static __device__ __forceinline__ float fast_tanh(float x) {
#if __has_builtin(__builtin_amdgcn_tanhf)
    return __builtin_amdgcn_tanhf(x);
#elif __has_builtin(__builtin_amdgcn_tanh_f32)
    return __builtin_amdgcn_tanh_f32(x);
#else
    return tanhf(x);
#endif
}

static __device__ __forceinline__ float sigf(float x) {
    return 0.5f * fast_tanh(0.5f * x) + 0.5f;
}

static __device__ __forceinline__ v16bf cat16(v8bf lo, v8bf hi) {
    return __builtin_shufflevector(lo, hi, 0,1,2,3,4,5,6,7,8,9,10,11,12,13,14,15);
}

static __device__ __forceinline__ unsigned pack_bf16x2(float a, float b) {
    union { __bf16 h[2]; unsigned u; } x;
    x.h[0] = (__bf16)a; x.h[1] = (__bf16)b;
    return x.u;
}

static __device__ __forceinline__ void wait_async0() {
#if USE_ASYNC_LDS
#if __has_builtin(__builtin_amdgcn_s_wait_asynccnt)
    __builtin_amdgcn_s_wait_asynccnt(0);
#else
    asm volatile("s_wait_asynccnt 0x0" ::: "memory");
#endif
#endif
}

// copy 32 bytes (16 bf16) global -> LDS, either async or via registers
static __device__ __forceinline__ void stage32(const __bf16* gsrc, __bf16* ldst) {
#if USE_ASYNC_LDS
    __builtin_amdgcn_global_load_async_to_lds_b128(
        (AS1 vsi4*)(gsrc),     (AS3 vsi4*)(ldst),     0, 0);
    __builtin_amdgcn_global_load_async_to_lds_b128(
        (AS1 vsi4*)(gsrc + 8), (AS3 vsi4*)(ldst + 8), 0, 0);
#else
    uint4 r0 = *(const uint4*)(gsrc);
    uint4 r1 = *(const uint4*)(gsrc + 8);
    *(uint4*)(ldst)     = r0;
    *(uint4*)(ldst + 8) = r1;
#endif
}

// ---------------------------------------------------------------------------
// Prep kernels
// ---------------------------------------------------------------------------

// Embedding gather: Xe[l*B+b][0:64] = bf16(emb[input[l,b]][0:50]) zero-padded
__global__ __launch_bounds__(256)
void k_prep_emb(const int* __restrict__ inp, const float* __restrict__ emb,
                __bf16* __restrict__ xe) {
    int p = blockIdx.x * 256 + threadIdx.x;          // 0 .. L*B-1
    if (p >= LL * BB) return;
    int v = inp[p];
    const float* e = emb + (size_t)v * EE;
    unsigned* dst = (unsigned*)(xe + (size_t)p * 64);
    #pragma unroll
    for (int j = 0; j < 32; ++j) {
        float a = (2 * j     < EE) ? e[2 * j]     : 0.0f;
        float b = (2 * j + 1 < EE) ? e[2 * j + 1] : 0.0f;
        dst[j] = pack_bf16x2(a, b);
    }
}

// Weight repack: packed row n = ntile*128 + gate*32 + ulocal  <->
// original row = gate*512 + (ntile*32 + ulocal).  Cols: [W_hh(512)|W_ih(50)|0].
__global__ __launch_bounds__(256)
void k_prep_w(const float* __restrict__ W_ih, const float* __restrict__ W_hh,
              const float* __restrict__ b_ih, const float* __restrict__ b_hh,
              __bf16* __restrict__ Wc, float* __restrict__ bpk) {
    int p = blockIdx.x * 256 + threadIdx.x;          // 0 .. 2048*576-1
    if (p >= FOURH * KCAT) return;
    int n = p / KCAT, k = p % KCAT;
    int gate = (n % 128) / 32;
    int unit = (n / 128) * 32 + (n % 32);
    int orig = gate * HH + unit;
    float val;
    if (k < HH)            val = W_hh[(size_t)orig * HH + k];
    else if (k < HH + EE)  val = W_ih[(size_t)orig * EE + (k - HH)];
    else                   val = 0.0f;
    Wc[p] = (__bf16)val;
    if (k == 0) bpk[n] = b_ih[orig] + b_hh[orig];
}

// W_out (V=128, H=512) row-major -> bf16 (already the N x K layout we want)
__global__ __launch_bounds__(256)
void k_prep_wout(const float* __restrict__ W_out, __bf16* __restrict__ Wo) {
    int p = blockIdx.x * 256 + threadIdx.x;
    if (p >= VV * HH) return;
    Wo[p] = (__bf16)W_out[p];
}

// h0 -> bf16 state buffer, c0 -> f32 state buffer
__global__ __launch_bounds__(256)
void k_prep_state(const float* __restrict__ h0, const float* __restrict__ c0,
                  __bf16* __restrict__ h, float* __restrict__ c) {
    int p = blockIdx.x * 256 + threadIdx.x;
    if (p >= BB * HH) return;
    h[p] = (__bf16)h0[p];
    c[p] = c0[p];
}

// ---------------------------------------------------------------------------
// Fused recurrent step: gates = [h|x] @ Wc^T + b; cell update in epilogue.
// Grid: (64 M-tiles, 16 N-tiles), block 256 = 8 wave32s.
// Wave w computes rows m0+16w..+15, all 128 packed cols (8 WMMA fragments).
// Double-buffered LDS, async global->LDS staging overlapped with WMMAs,
// B-fragment ds_loads software-pipelined one fragment ahead.
// ---------------------------------------------------------------------------
__global__ __launch_bounds__(256)
void k_lstm_step(const __bf16* __restrict__ h_in, const __bf16* __restrict__ xe_l,
                 const __bf16* __restrict__ Wc, const float* __restrict__ bpk,
                 float* __restrict__ c, __bf16* __restrict__ h_out,
                 float* __restrict__ hT, float* __restrict__ cT) {
    __shared__ __bf16 Als[2][128 * LDST];
    __shared__ __bf16 Bls[2][128 * LDST];

    const int tid  = threadIdx.x;
    const int wave = tid >> 5;
    const int lane = tid & 31;
    const int m0   = blockIdx.x * 128;
    const int n0   = blockIdx.y * 128;

    v8f acc[8] = {};

    const int srow = tid >> 1;          // staging row 0..127
    const int scol = (tid & 1) * 16;    // 0 or 16

    const int lc   = lane & 15;
    const int arow = wave * 16 + lc;
    const int akb  = (lane < 16) ? 0 : 8;    // 16-bit A: K 0-7/16-23 vs 8-15/24-31
    const int bkb  = (lane < 16) ? 0 : 8;    // 16-bit B: K 0-7/16-23 vs 8-15/24-31

    #define LSTM_STAGE(BUF, KC) do {                                           \
        const int k0_ = (KC) * 32;                                             \
        const __bf16* asrc_ = (k0_ < HH)                                       \
            ? h_in + (size_t)(m0 + srow) * HH + (k0_ + scol)                   \
            : xe_l + (size_t)(m0 + srow) * 64 + (k0_ - HH + scol);             \
        stage32(asrc_, Als[BUF] + srow * LDST + scol);                         \
        const __bf16* bsrc_ = Wc + (size_t)(n0 + srow) * KCAT + (k0_ + scol);  \
        stage32(bsrc_, Bls[BUF] + srow * LDST + scol);                         \
    } while (0)

    LSTM_STAGE(0, 0);
    wait_async0();
    __syncthreads();

    #pragma unroll 1
    for (int kc = 0; kc < 18; ++kc) {
        const int cur = kc & 1;
        if (kc + 1 < 18) LSTM_STAGE(cur ^ 1, kc + 1);   // copy next chunk in bg

        const __bf16* Ab = Als[cur];
        const __bf16* Bb = Bls[cur];
        v8bf alo = *(const v8bf*)(Ab + arow * LDST + akb);
        v8bf ahi = *(const v8bf*)(Ab + arow * LDST + akb + 16);
        v16bf a  = cat16(alo, ahi);
        // pipeline B-fragment loads one ahead of the WMMA consuming them
        const __bf16* bp0 = Bb + lc * LDST + bkb;
        v8bf blo = *(const v8bf*)(bp0);
        v8bf bhi = *(const v8bf*)(bp0 + 16);
        #pragma unroll
        for (int t = 0; t < 8; ++t) {
            v16bf b = cat16(blo, bhi);
            if (t < 7) {
                const __bf16* bp = Bb + ((t + 1) * 16 + lc) * LDST + bkb;
                blo = *(const v8bf*)(bp);
                bhi = *(const v8bf*)(bp + 16);
            }
            acc[t] = __builtin_amdgcn_wmma_f32_16x16x32_bf16(
                false, a, false, b, (short)0, acc[t], false, false);
        }

        wait_async0();                                   // our async copies done
        __syncthreads();                                 // everyone's done
    }
    #undef LSTM_STAGE

    // Epilogue: fragments {p, p+2, p+4, p+6} hold i/f/g/o for the same (m,unit)
    const int mrow_base = m0 + wave * 16 + ((lane < 16) ? 0 : 8);
    #pragma unroll
    for (int p = 0; p < 2; ++p) {
        const int ucol = p * 16 + lc;                 // unit-local 0..31
        const float bi = bpk[n0 +  0 + ucol];
        const float bf = bpk[n0 + 32 + ucol];
        const float bg = bpk[n0 + 64 + ucol];
        const float bo = bpk[n0 + 96 + ucol];
        const int unit = (n0 / 128) * 32 + ucol;      // global hidden index
        #pragma unroll
        for (int r = 0; r < 8; ++r) {
            const int m = mrow_base + r;
            const size_t idx = (size_t)m * HH + unit;
            const float iv = acc[p    ][r] + bi;
            const float fv = acc[p + 2][r] + bf;
            const float gv = acc[p + 4][r] + bg;
            const float ov = acc[p + 6][r] + bo;
            const float co = c[idx];
            const float cn = sigf(fv) * co + sigf(iv) * fast_tanh(gv);
            const float hn = sigf(ov) * fast_tanh(cn);
            c[idx] = cn;
            h_out[idx] = (__bf16)hn;
            if (hT) { hT[idx] = hn; cT[idx] = cn; }
        }
    }
}

// ---------------------------------------------------------------------------
// Output projection: scores(B,128) = h(B,512) @ W_out^T + b_out
// Grid: 64 M-tiles, single N tile (V=128). Same fragment scheme, K=512.
// ---------------------------------------------------------------------------
__global__ __launch_bounds__(256)
void k_proj_step(const __bf16* __restrict__ h, const __bf16* __restrict__ Wo,
                 const float* __restrict__ b_out, float* __restrict__ out) {
    __shared__ __bf16 Als[2][128 * LDST];
    __shared__ __bf16 Bls[2][128 * LDST];

    const int tid  = threadIdx.x;
    const int wave = tid >> 5;
    const int lane = tid & 31;
    const int m0   = blockIdx.x * 128;

    v8f acc[8] = {};

    const int srow = tid >> 1;
    const int scol = (tid & 1) * 16;
    const int lc   = lane & 15;
    const int arow = wave * 16 + lc;
    const int akb  = (lane < 16) ? 0 : 8;
    const int bkb  = (lane < 16) ? 0 : 8;

    #define PROJ_STAGE(BUF, KC) do {                                           \
        const int k0_ = (KC) * 32;                                             \
        const __bf16* asrc_ = h + (size_t)(m0 + srow) * HH + (k0_ + scol);     \
        stage32(asrc_, Als[BUF] + srow * LDST + scol);                         \
        const __bf16* bsrc_ = Wo + (size_t)srow * HH + (k0_ + scol);           \
        stage32(bsrc_, Bls[BUF] + srow * LDST + scol);                         \
    } while (0)

    PROJ_STAGE(0, 0);
    wait_async0();
    __syncthreads();

    #pragma unroll 1
    for (int kc = 0; kc < 16; ++kc) {
        const int cur = kc & 1;
        if (kc + 1 < 16) PROJ_STAGE(cur ^ 1, kc + 1);

        const __bf16* Ab = Als[cur];
        const __bf16* Bb = Bls[cur];
        v8bf alo = *(const v8bf*)(Ab + arow * LDST + akb);
        v8bf ahi = *(const v8bf*)(Ab + arow * LDST + akb + 16);
        v16bf a  = cat16(alo, ahi);
        const __bf16* bp0 = Bb + lc * LDST + bkb;
        v8bf blo = *(const v8bf*)(bp0);
        v8bf bhi = *(const v8bf*)(bp0 + 16);
        #pragma unroll
        for (int t = 0; t < 8; ++t) {
            v16bf b = cat16(blo, bhi);
            if (t < 7) {
                const __bf16* bp = Bb + ((t + 1) * 16 + lc) * LDST + bkb;
                blo = *(const v8bf*)(bp);
                bhi = *(const v8bf*)(bp + 16);
            }
            acc[t] = __builtin_amdgcn_wmma_f32_16x16x32_bf16(
                false, a, false, b, (short)0, acc[t], false, false);
        }

        wait_async0();
        __syncthreads();
    }
    #undef PROJ_STAGE

    const int mrow_base = m0 + wave * 16 + ((lane < 16) ? 0 : 8);
    #pragma unroll
    for (int t = 0; t < 8; ++t) {
        const int n = t * 16 + lc;
        const float bo = b_out[n];
        #pragma unroll
        for (int r = 0; r < 8; ++r) {
            const int m = mrow_base + r;
            out[(size_t)m * VV + n] = acc[t][r] + bo;
        }
    }
}

// ---------------------------------------------------------------------------
// Host launcher
// ---------------------------------------------------------------------------
extern "C" void kernel_launch(void* const* d_in, const int* in_sizes, int n_in,
                              void* d_out, int out_size, void* d_ws, size_t ws_size,
                              hipStream_t stream) {
    (void)in_sizes; (void)n_in; (void)out_size; (void)ws_size;
    const int*   inp   = (const int*)  d_in[0];
    const float* h0    = (const float*)d_in[1];
    const float* c0    = (const float*)d_in[2];
    const float* emb   = (const float*)d_in[3];
    const float* W_ih  = (const float*)d_in[4];
    const float* W_hh  = (const float*)d_in[5];
    const float* b_ih  = (const float*)d_in[6];
    const float* b_hh  = (const float*)d_in[7];
    const float* W_out = (const float*)d_in[8];
    const float* b_out = (const float*)d_in[9];

    // workspace carve-up (256B aligned slices)
    char* ws = (char*)d_ws;
    size_t off = 0;
    auto take = [&](size_t bytes) -> void* {
        void* p = ws + off;
        off = (off + bytes + 255) & ~(size_t)255;
        return p;
    };
    __bf16* Xe   = (__bf16*)take((size_t)LL * BB * 64 * 2);       // 50.3 MB
    __bf16* Wc   = (__bf16*)take((size_t)FOURH * KCAT * 2);       //  2.4 MB
    float*  bpk  = (float*) take((size_t)FOURH * 4);
    __bf16* Wo   = (__bf16*)take((size_t)VV * HH * 2);
    __bf16* hA   = (__bf16*)take((size_t)BB * HH * 2);            //  8.4 MB
    __bf16* hB   = (__bf16*)take((size_t)BB * HH * 2);            //  8.4 MB
    float*  cbuf = (float*) take((size_t)BB * HH * 4);            // 16.8 MB

    float* scores = (float*)d_out;                 // (L, B, V)
    float* hT = scores + (size_t)LL * BB * VV;     // (B, H)
    float* cT = hT + (size_t)BB * HH;              // (B, H)

    // prep
    k_prep_emb  <<<(LL * BB + 255) / 256, 256, 0, stream>>>(inp, emb, Xe);
    k_prep_w    <<<(FOURH * KCAT + 255) / 256, 256, 0, stream>>>(W_ih, W_hh, b_ih, b_hh, Wc, bpk);
    k_prep_wout <<<(VV * HH + 255) / 256, 256, 0, stream>>>(W_out, Wo);
    k_prep_state<<<(BB * HH + 255) / 256, 256, 0, stream>>>(h0, c0, hA, cbuf);

    // recurrent loop: one fused GEMM+cell kernel and one projection per step
    __bf16* hin = hA;
    __bf16* hout = hB;
    for (int l = 0; l < LL; ++l) {
        const bool last = (l == LL - 1);
        k_lstm_step<<<dim3(BB / 128, FOURH / 128), 256, 0, stream>>>(
            hin, Xe + (size_t)l * BB * 64, Wc, bpk, cbuf, hout,
            last ? hT : nullptr, last ? cT : nullptr);
        k_proj_step<<<dim3(BB / 128), 256, 0, stream>>>(
            hout, Wo, b_out, scores + (size_t)l * BB * VV);
        __bf16* tmp = hin; hin = hout; hout = tmp;
    }
}
